// PPNet_GNN_brach_noise_unit_55465207660926
// MI455X (gfx1250) — compile-verified, matrix operand order checked by reading, not run on hardware
//
#include <hip/hip_runtime.h>

// ---------------------------------------------------------------------------
// PPNet-style point-cloud GNN for MI455X (gfx1250, wave32, WMMA).
//
// GEMMs run through v_wmma_f32_16x16x32_f16 with:
//  - A tile: guard-free float4 global loads (all activation matrices are kept
//    zero-padded to 32-column strides), f32->f16 convert, packed 4-half
//    ds_store_b64; fragments read as two ds_load_b128 per lane (ISA 16-bit
//    fragment layout = two contiguous 8-half runs per lane).
//  - B tile: weights pre-packed to transposed zero-padded f16 Wp[CoutP][CinP];
//    staged global->LDS with one global_load_async_to_lds_b128 per thread
//    (ASYNCcnt + s_wait_asynccnt); all four B fragments preloaded, then four
//    WMMAs issued back-to-back with the A fragment reused.
// Block tile 128x64 = 8 wave32 waves x (16 rows x 64 cols).
//
// Input layout assumption (compile-only; documented):
//   d_in[0] point_cloud (B,N,3) f32     d_in[1] feat (B,64,N) f32
//   d_in[2] noise3 (B,3,N) f32          d_in[3] noise16 (B,16,N) f32
//   d_in[4..] params dict flattened in JAX pytree (sorted-key) order.
// ---------------------------------------------------------------------------

#define BQ   2
#define NPTS 8192
#define KNN  8
#define MPTS (BQ * NPTS)    // 16384 points
#define EDG  (MPTS * KNN)   // 131072 edges

typedef __attribute__((ext_vector_type(16))) _Float16 v16h;
typedef __attribute__((ext_vector_type(8)))  _Float16 v8h;
typedef __attribute__((ext_vector_type(4)))  _Float16 v4h;
typedef __attribute__((ext_vector_type(8)))  float    v8f;

// ---------------------------------------------------------------------------
// Kernel 1: pairwise distances + top-8 nearest / top-8 farthest + covariance
// + geo edge features. One thread per point; point cloud streamed through LDS
// in 1024-point chunks (12 KB). Top-8 lists kept fully in registers via an
// unrolled bubble-insert (no dynamic indexing -> no scratch).
// x0 is written with stride 32 (cols 15..31 zeroed) for guard-free GEMM loads.
// ---------------------------------------------------------------------------
__global__ __launch_bounds__(256) void prep_knn_kernel(
    const float* __restrict__ pc,      // (B,N,3)
    const float* __restrict__ noise3,  // (B,3,N)
    float* __restrict__ x0,            // (M,32) = [pc | cov9 | noise3 | 0...]
    int*   __restrict__ sid1,          // (E) nearest flat ids
    int*   __restrict__ sid2,          // (E) farthest flat ids
    float* __restrict__ pce,           // (E,3)
    int*   __restrict__ bid,           // (E,3)
    float* __restrict__ dnorm)         // (E)
{
  __shared__ float sx[1024], sy[1024], sz[1024];
  const int tid = threadIdx.x;
  const int m   = blockIdx.x * 256 + tid;   // block never straddles a batch
  const int b   = m / NPTS;
  const int n   = m - b * NPTS;
  const int bbase = b * NPTS;

  const float xn = pc[m * 3 + 0], yn = pc[m * 3 + 1], zn = pc[m * 3 + 2];
  const float nn = xn * xn + yn * yn + zn * zn;

  float bv[KNN]; int bi_[KNN];   // largest pair values  (nearest, self first)
  float wv[KNN]; int wi_[KNN];   // smallest pair values (farthest)
#pragma unroll
  for (int j = 0; j < KNN; ++j) { bv[j] = -3.4e38f; bi_[j] = 0; wv[j] = 3.4e38f; wi_[j] = 0; }

  for (int ch = 0; ch < NPTS / 1024; ++ch) {
    __syncthreads();
    for (int i = tid; i < 1024; i += 256) {
      int g = (bbase + ch * 1024 + i) * 3;
      sx[i] = pc[g]; sy[i] = pc[g + 1]; sz[i] = pc[g + 2];
    }
    __syncthreads();
    for (int i = 0; i < 1024; ++i) {
      float qx = sx[i], qy = sy[i], qz = sz[i];
      float v = 2.f * (xn * qx + yn * qy + zn * qz) - nn - (qx * qx + qy * qy + qz * qz);
      int idx = ch * 1024 + i;
      float cv = v; int ci = idx;
#pragma unroll
      for (int j = 0; j < KNN; ++j)
        if (cv > bv[j]) { float tf = bv[j]; int ti = bi_[j]; bv[j] = cv; bi_[j] = ci; cv = tf; ci = ti; }
      cv = v; ci = idx;
#pragma unroll
      for (int j = 0; j < KNN; ++j)
        if (cv < wv[j]) { float tf = wv[j]; int ti = wi_[j]; wv[j] = cv; wi_[j] = ci; cv = tf; ci = ti; }
    }
  }

  // --- covariance of nearest-8 neighbor coords (mean-centered) -------------
  float qx[KNN], qy[KNN], qz[KNN];
  float mx = 0.f, my = 0.f, mz = 0.f;
#pragma unroll
  for (int k = 0; k < KNN; ++k) {
    int g = (bbase + bi_[k]) * 3;
    qx[k] = pc[g]; qy[k] = pc[g + 1]; qz[k] = pc[g + 2];
    mx += qx[k]; my += qy[k]; mz += qz[k];
    sid1[m * KNN + k] = bbase + bi_[k];
  }
  mx *= 0.125f; my *= 0.125f; mz *= 0.125f;
  float c00 = 0, c01 = 0, c02 = 0, c11 = 0, c12 = 0, c22 = 0;
#pragma unroll
  for (int k = 0; k < KNN; ++k) {
    float dx = qx[k] - mx, dy = qy[k] - my, dz = qz[k] - mz;
    c00 += dx * dx; c01 += dx * dy; c02 += dx * dz;
    c11 += dy * dy; c12 += dy * dz; c22 += dz * dz;
  }
  float* xr = x0 + (size_t)m * 32;
  xr[0] = xn; xr[1] = yn; xr[2] = zn;
  xr[3] = c00; xr[4] = c01; xr[5] = c02;
  xr[6] = c01; xr[7] = c11; xr[8] = c12;
  xr[9] = c02; xr[10] = c12; xr[11] = c22;
  xr[12] = noise3[(b * 3 + 0) * NPTS + n];
  xr[13] = noise3[(b * 3 + 1) * NPTS + n];
  xr[14] = noise3[(b * 3 + 2) * NPTS + n];
#pragma unroll
  for (int j = 15; j < 32; ++j) xr[j] = 0.f;

  // --- farthest-8 geo features --------------------------------------------
  float dis[KNN]; float pr = 0.f;
#pragma unroll
  for (int k = 0; k < KNN; ++k) {
    int s = bbase + wi_[k];
    sid2[m * KNN + k] = s;
    float dx = pc[s * 3 + 0] - xn, dy = pc[s * 3 + 1] - yn, dz = pc[s * 3 + 2] - zn;
    float d = sqrtf(dx * dx + dy * dy + dz * dz);
    d = fmaxf(d, 1e-16f);
    dis[k] = d; pr = fmaxf(pr, d);
    int e3 = (m * KNN + k) * 3;
    float c0 = cosf(dx / d), c1 = cosf(dy / d), c2 = cosf(dz / d);
    pce[e3 + 0] = c0 * c0; pce[e3 + 1] = c1 * c1; pce[e3 + 2] = c2 * c2;
    bid[e3 + 0] = (dx > 0.f ? 1 : 0) + 0;
    bid[e3 + 1] = (dy > 0.f ? 1 : 0) + 2;
    bid[e3 + 2] = (dz > 0.f ? 1 : 0) + 4;
  }
  pr *= 1.1f;
  float pd[KNN]; float sum = 0.f;
#pragma unroll
  for (int k = 0; k < KNN; ++k) { float d = pr - dis[k]; pd[k] = d * d; sum += pd[k]; }
  sum = fmaxf(sum, 1e-16f);
#pragma unroll
  for (int k = 0; k < KNN; ++k) dnorm[m * KNN + k] = pd[k] / sum;
}

// ---------------------------------------------------------------------------
// Pack f32 weights (Cin,Cout) -> transposed zero-padded f16 Wp (CoutP,CinP)
// ---------------------------------------------------------------------------
__global__ void pack_w_f16_kernel(const float* __restrict__ W, _Float16* __restrict__ Wp,
                                  int Cin, int CinP, int Cout, int CoutP)
{
  int t = blockIdx.x * 256 + threadIdx.x;
  if (t >= CoutP * CinP) return;
  int co = t / CinP, k = t - co * CinP;
  float v = (co < Cout && k < Cin) ? W[(size_t)k * Cout + co] : 0.f;
  Wp[t] = (_Float16)v;
}

// ---------------------------------------------------------------------------
// WMMA GEMM: Y(Mr,Cout) = X[gather](Mr,CinP) * Wp^T + bias (+ addend)
// X has stride CinP (32-multiple, zero-padded). Mr is a 128-multiple.
// ---------------------------------------------------------------------------
__global__ __launch_bounds__(256) void gemm_wmma_kernel(
    const float* __restrict__ X, const int* __restrict__ gather,
    const _Float16* __restrict__ Wp, const float* __restrict__ bias,
    const float* __restrict__ addend, float* __restrict__ Y,
    int CinP, int Cout)
{
  __shared__ _Float16 As[128][40];   // padded rows (80 B) keep 16-B alignment
  __shared__ _Float16 Bt[64][40];    // B tile stored transposed: Bt[col][k]
  const int tid  = threadIdx.x;
  const int lane = tid & 31;         // wave32
  const int wv   = tid >> 5;         // 8 waves
  const int mrow = lane & 15;
  const int half = lane >> 4;
  const int rowBase = blockIdx.x * 128;
  const int colBase = blockIdx.y * 64;

  // per-thread A-stage coordinates: 1024 chunks of 4 floats, 4 per thread
  const int ar  = tid >> 3;          // base row (0..31), +32 per iteration
  const int akq = (tid & 7) * 4;     // 4-float chunk within 32 cols

  // per-thread async-copy coordinates for the B tile (64 rows x 64 B)
  const int brow = tid >> 2;         // 0..63 : output-channel row of Bt
  const int bchk = tid & 3;          // 0..3  : 16-byte chunk within 32 halves
  const unsigned ldsB = (unsigned)(uintptr_t)&Bt[brow][bchk * 8];
  const unsigned long long wbase = (unsigned long long)(uintptr_t)Wp;

  v8f acc[4] = {};
  const int ksteps = CinP >> 5;

  for (int ks = 0; ks < ksteps; ++ks) {
    __syncthreads();
    // ---- stage A tile (128x32): float4 loads, packed 4-half stores --------
#pragma unroll
    for (int it = 0; it < 4; ++it) {
      int r = ar + it * 32;
      int gr = rowBase + r;
      int src = gather ? gather[gr] : gr;
      const float4 v = *(const float4*)(X + (size_t)src * CinP + ks * 32 + akq);
      v4h p;
      p[0] = (_Float16)v.x; p[1] = (_Float16)v.y;
      p[2] = (_Float16)v.z; p[3] = (_Float16)v.w;
      *(v4h*)&As[r][akq] = p;
    }
    // ---- stage B tile: one async b128 copy per thread (global -> LDS) -----
    {
      unsigned voff = (unsigned)((((colBase + brow) * CinP) + ks * 32 + bchk * 8) * 2);
      asm volatile("global_load_async_to_lds_b128 %0, %1, %2"
                   :: "v"(ldsB), "v"(voff), "s"(wbase) : "memory");
#if __has_builtin(__builtin_amdgcn_s_wait_asynccnt)
      __builtin_amdgcn_s_wait_asynccnt(0);
#else
      asm volatile("s_wait_asynccnt 0x0" ::: "memory");
#endif
    }
    if (ks + 1 < ksteps)
      __builtin_prefetch(&X[(size_t)rowBase * CinP + ks * 32 + 32], 0, 0);
    __syncthreads();

    // ---- fragments: two ds_load_b128 each; preload B then back-to-back WMMA
    const _Float16* arow = &As[wv * 16 + mrow][0];
    v8h alo = *(const v8h*)(arow + half * 8);
    v8h ahi = *(const v8h*)(arow + 16 + half * 8);
    v16h af = __builtin_shufflevector(alo, ahi, 0, 1, 2, 3, 4, 5, 6, 7,
                                      8, 9, 10, 11, 12, 13, 14, 15);
    v16h bf[4];
#pragma unroll
    for (int t = 0; t < 4; ++t) {
      const _Float16* brp = &Bt[t * 16 + mrow][0];
      v8h blo = *(const v8h*)(brp + half * 8);
      v8h bhi = *(const v8h*)(brp + 16 + half * 8);
      bf[t] = __builtin_shufflevector(blo, bhi, 0, 1, 2, 3, 4, 5, 6, 7,
                                      8, 9, 10, 11, 12, 13, 14, 15);
    }
#pragma unroll
    for (int t = 0; t < 4; ++t)
      acc[t] = __builtin_amdgcn_wmma_f32_16x16x32_f16(
          false, af, false, bf[t], (short)0, acc[t], false, false);
  }

  // ---- epilogue: C/D layout -> lane col = lane&15, VGPR r -> row r+8*half --
#pragma unroll
  for (int t = 0; t < 4; ++t) {
    int col = colBase + t * 16 + mrow;
    if (col >= Cout) continue;
    float bz = bias ? bias[col] : 0.f;
#pragma unroll
    for (int r = 0; r < 8; ++r) {
      int row = rowBase + wv * 16 + r + 8 * half;
      float v = acc[t][r] + bz;
      if (addend) v += addend[(size_t)row * Cout + col];
      Y[(size_t)row * Cout + col] = v;
    }
  }
}

// ---------------------------------------------------------------------------
// BatchNorm (training-mode batch stats) and activation
// ---------------------------------------------------------------------------
__global__ void bnstat_kernel(const float* __restrict__ X, int Mr, int C,
                              float* __restrict__ mean, float* __restrict__ var)
{
  __shared__ float ss[256], sq[256];
  int c = blockIdx.x, tid = threadIdx.x;
  float s = 0.f, q = 0.f;
  for (int r = tid; r < Mr; r += 256) {
    float v = X[(size_t)r * C + c];
    s += v; q += v * v;
  }
  ss[tid] = s; sq[tid] = q; __syncthreads();
  for (int st = 128; st > 0; st >>= 1) {
    if (tid < st) { ss[tid] += ss[tid + st]; sq[tid] += sq[tid + st]; }
    __syncthreads();
  }
  if (tid == 0) {
    float mu = ss[0] / (float)Mr;
    mean[c] = mu;
    var[c]  = sq[0] / (float)Mr - mu * mu;
  }
}

// act: 0=none, 1=leaky(0.2), 2=relu, 3=sigmoid
__global__ void bnapply_kernel(float* __restrict__ X, int Mr, int C,
                               const float* __restrict__ mean, const float* __restrict__ var,
                               const float* __restrict__ g, const float* __restrict__ be, int act)
{
  int t = blockIdx.x * 256 + threadIdx.x;
  if (t >= Mr * C) return;
  int c = t % C;
  float v = X[t];
  v = g[c] * (v - mean[c]) * rsqrtf(var[c] + 1e-5f) + be[c];
  if (act == 1)      v = v > 0.f ? v : 0.2f * v;
  else if (act == 2) v = fmaxf(v, 0.f);
  else if (act == 3) v = 1.f / (1.f + expf(-v));
  X[t] = v;
}

// max-pool over K neighbors: (M*K, C) -> (M, C)
__global__ void pool_max_kernel(const float* __restrict__ msg, float* __restrict__ out, int C)
{
  int t = blockIdx.x * 256 + threadIdx.x;
  if (t >= MPTS * C) return;
  int m = t / C, c = t - m * C;
  float v = -3.4e38f;
  for (int k = 0; k < KNN; ++k) v = fmaxf(v, msg[(size_t)(m * KNN + k) * C + c]);
  out[t] = v;
}

// geoconv: select bins by bid, weight by pce, sum axes, weight by dnorm, sum K
__global__ void geo_agg_kernel(const float* __restrict__ x6, const float* __restrict__ pce,
                               const int* __restrict__ bid, const float* __restrict__ dn,
                               float* __restrict__ agg, int hid)
{
  int t = blockIdx.x * 256 + threadIdx.x;
  if (t >= MPTS * hid) return;
  int m = t / hid, h = t - m * hid;
  int W6 = 6 * hid;
  float acc = 0.f;
  for (int k = 0; k < KNN; ++k) {
    int e = m * KNN + k;
    float msg = 0.f;
#pragma unroll
    for (int ax = 0; ax < 3; ++ax)
      msg += pce[e * 3 + ax] * x6[(size_t)e * W6 + bid[e * 3 + ax] * hid + h];
    acc += msg * dn[e];
  }
  agg[t] = acc;
}

// Wbins (6,cin,hid) -> Wt (cin, 6*hid), column j = m*hid + h
__global__ void pack_wbins_kernel(const float* __restrict__ Wb, float* __restrict__ Wt,
                                  int cin, int hid)
{
  int W6 = 6 * hid;
  int t = blockIdx.x * 256 + threadIdx.x;
  if (t >= cin * W6) return;
  int c = t / W6, j = t - c * W6;
  int m = j / hid, h = j - m * hid;
  Wt[t] = Wb[(size_t)(m * cin + c) * hid + h];
}

// (B,C,N) -> (M,C)
__global__ void tr_bcn_kernel(const float* __restrict__ src, float* __restrict__ dst, int C)
{
  int t = blockIdx.x * 256 + threadIdx.x;
  if (t >= MPTS * C) return;
  int m = t / C, c = t - m * C;
  int b = m / NPTS, n = m - b * NPTS;
  dst[t] = src[(size_t)(b * C + c) * NPTS + n];
}

// row-wise concat of up to three (M,*) matrices into stride Wdst (zero tail)
__global__ void concat3_kernel(float* __restrict__ dst, int Wdst,
                               const float* __restrict__ a, int wa,
                               const float* __restrict__ b_, int wb,
                               const float* __restrict__ c_, int wc)
{
  int t = blockIdx.x * 256 + threadIdx.x;
  if (t >= MPTS * Wdst) return;
  int m = t / Wdst, j = t - m * Wdst;
  float v = 0.f;
  if (j < wa)                 v = a[(size_t)m * wa + j];
  else if (j < wa + wb)       v = b_[(size_t)m * wb + (j - wa)];
  else if (j < wa + wb + wc)  v = c_[(size_t)m * wc + (j - wa - wb)];
  dst[t] = v;
}

// cat_h = [l0 | r * featT]
__global__ void gru_cath_kernel(const float* __restrict__ l0, const float* __restrict__ r,
                                const float* __restrict__ ft, float* __restrict__ dst)
{
  int t = blockIdx.x * 256 + threadIdx.x;
  if (t >= MPTS * 128) return;
  int m = t / 128, j = t - m * 128;
  dst[t] = (j < 64) ? l0[m * 64 + j] : r[m * 64 + j - 64] * ft[m * 64 + j - 64];
}

// feat2 = (1-z)*l0 + z*h_hat
__global__ void gru_comb_kernel(const float* __restrict__ z, const float* __restrict__ l0,
                                const float* __restrict__ hh, float* __restrict__ out)
{
  int t = blockIdx.x * 256 + threadIdx.x;
  if (t >= MPTS * 64) return;
  out[t] = (1.f - z[t]) * l0[t] + z[t] * hh[t];
}

// out0 = seg + pc^T (B,3,N), out1 = seg (B,3,N), concatenated flat
__global__ void final_out_kernel(const float* __restrict__ seg, const float* __restrict__ pc,
                                 float* __restrict__ out)
{
  int t = blockIdx.x * 256 + threadIdx.x;
  if (t >= BQ * 3 * NPTS) return;
  int b = t / (3 * NPTS);
  int rem = t - b * 3 * NPTS;
  int c = rem / NPTS, n = rem - c * NPTS;
  float s = seg[(size_t)(b * NPTS + n) * 3 + c];
  out[t] = s + pc[(size_t)(b * NPTS + n) * 3 + c];
  out[BQ * 3 * NPTS + t] = s;
}

// ---------------------------------------------------------------------------
// Host-side helpers
// ---------------------------------------------------------------------------
static inline unsigned gfor(size_t n) { return (unsigned)((n + 255) / 256); }

// CinW = real weight rows; CinP = X stride (32-multiple, >= CinW, zero-padded)
static inline void run_gemm(const float* X, const int* gather, const float* W,
                            const float* bias, const float* addend, float* Y,
                            int Mr, int CinW, int CinP, int Cout,
                            _Float16* Wp, hipStream_t s)
{
  int CoutP = (Cout + 63) & ~63;
  pack_w_f16_kernel<<<gfor((size_t)CoutP * CinP), 256, 0, s>>>(W, Wp, CinW, CinP, Cout, CoutP);
  dim3 g((unsigned)(Mr / 128), (unsigned)(CoutP / 64));
  gemm_wmma_kernel<<<g, 256, 0, s>>>(X, gather, Wp, bias, addend, Y, CinP, Cout);
}
static inline void run_bn(float* X, int Mr, int C, float* mean, float* var,
                          const float* g_, const float* be, int act, hipStream_t s)
{
  bnstat_kernel<<<(unsigned)C, 256, 0, s>>>(X, Mr, C, mean, var);
  int tot = Mr * C;
  bnapply_kernel<<<gfor((size_t)tot), 256, 0, s>>>(X, Mr, C, mean, var, g_, be, act);
}

extern "C" void kernel_launch(void* const* d_in, const int* in_sizes, int n_in,
                              void* d_out, int out_size, void* d_ws, size_t ws_size,
                              hipStream_t stream)
{
  (void)in_sizes; (void)n_in; (void)out_size; (void)ws_size;

  // ---- input indexing (sorted-key pytree order for params) ----------------
  const float* PC  = (const float*)d_in[0];
  const float* FEA = (const float*)d_in[1];
  const float* N3  = (const float*)d_in[2];
  const float* N16 = (const float*)d_in[3];
#define PAR(i) ((const float*)d_in[(i)])
  const int PB = 4;
  // CBL blocks (sorted: W, b, be, g)
  const int C1A = PB + 0, C1B = PB + 4, C1C = PB + 8, C2 = PB + 12;
  const int F1 = PB + 16, F2 = PB + 20, F3 = PB + 24;
  const int F4W = PB + 28, F4B = PB + 29;
  // GEO blocks (sorted: Wbins, Wout, Wself, bbins, be1, be2, bout, bself, g1, g2)
  const int G1 = PB + 30, G2 = PB + 40;
  const int PP1 = PB + 50, PP2 = PB + 54;
  const int S1 = PB + 58, S2W = PB + 62, S2B = PB + 63;
  const int UH = PB + 64, UR = PB + 68, UZ = PB + 72;

  // ---- workspace bump allocator -------------------------------------------
  char* wsb = (char*)d_ws;
  size_t off = 0;
  auto alloc = [&](size_t bytes) -> void* {
    void* p = wsb + off;
    off = (off + bytes + 255) & ~(size_t)255;
    return p;
  };
  int*      sid1  = (int*)     alloc((size_t)EDG * 4);
  int*      sid2  = (int*)     alloc((size_t)EDG * 4);
  int*      bidb  = (int*)     alloc((size_t)EDG * 3 * 4);
  float*    pceb  = (float*)   alloc((size_t)EDG * 3 * 4);
  float*    dnb   = (float*)   alloc((size_t)EDG * 4);
  float*    x0    = (float*)   alloc((size_t)MPTS * 32 * 4);   // padded 15->32
  float*    xA    = (float*)   alloc((size_t)MPTS * 640 * 4);
  float*    xB    = (float*)   alloc((size_t)MPTS * 640 * 4);
  float*    x3    = (float*)   alloc((size_t)MPTS * 64 * 4);
  float*    h2    = (float*)   alloc((size_t)MPTS * 256 * 4);
  float*    l3    = (float*)   alloc((size_t)MPTS * 128 * 4);
  float*    l4    = (float*)   alloc((size_t)MPTS * 256 * 4);
  float*    l0    = (float*)   alloc((size_t)MPTS * 64 * 4);
  float*    featT = (float*)   alloc((size_t)MPTS * 64 * 4);
  float*    n16T  = (float*)   alloc((size_t)MPTS * 16 * 4);
  float*    zb    = (float*)   alloc((size_t)MPTS * 64 * 4);
  float*    rb    = (float*)   alloc((size_t)MPTS * 64 * 4);
  float*    hhb   = (float*)   alloc((size_t)MPTS * 64 * 4);
  float*    agg   = (float*)   alloc((size_t)MPTS * 64 * 4);
  float*    Wt    = (float*)   alloc((size_t)128 * 384 * 4);
  _Float16* Wp    = (_Float16*)alloc((size_t)512 * 640 * 2);   // max packed weight
  float*    mean  = (float*)   alloc((size_t)1024 * 4);
  float*    var   = (float*)   alloc((size_t)1024 * 4);
  float*    seg   = (float*)   alloc((size_t)MPTS * 3 * 4);
  float*    BIG   = (float*)   alloc((size_t)EDG * 384 * 4);   // shared big arena

  // ---- stage 0: knn / topk / geo features ---------------------------------
  prep_knn_kernel<<<MPTS / 256, 256, 0, stream>>>(PC, N3, x0, sid1, sid2, pceb, bidb, dnb);

  // ---- c1a / c1b / c1c (15->64->64->64), leaky BN -------------------------
  run_gemm(x0, nullptr, PAR(C1A + 0), PAR(C1A + 1), nullptr, xA, MPTS, 15, 32, 64, Wp, stream);
  run_bn(xA, MPTS, 64, mean, var, PAR(C1A + 3), PAR(C1A + 2), 1, stream);
  run_gemm(xA, nullptr, PAR(C1B + 0), PAR(C1B + 1), nullptr, xB, MPTS, 64, 64, 64, Wp, stream);
  run_bn(xB, MPTS, 64, mean, var, PAR(C1B + 3), PAR(C1B + 2), 1, stream);
  run_gemm(xB, nullptr, PAR(C1C + 0), PAR(C1C + 1), nullptr, x3, MPTS, 64, 64, 64, Wp, stream);
  run_bn(x3, MPTS, 64, mean, var, PAR(C1C + 3), PAR(C1C + 2), 1, stream);

  // ---- pointplus pp1 / pp2 (gathered GEMM, K-maxpool, BN relu) ------------
  run_gemm(x3, sid1, PAR(PP1 + 0), PAR(PP1 + 1), nullptr, BIG, EDG, 64, 64, 128, Wp, stream);
  pool_max_kernel<<<gfor((size_t)MPTS * 128), 256, 0, stream>>>(BIG, xA, 128);
  run_bn(xA, MPTS, 128, mean, var, PAR(PP1 + 3), PAR(PP1 + 2), 2, stream);
  run_gemm(xA, sid1, PAR(PP2 + 0), PAR(PP2 + 1), nullptr, BIG, EDG, 128, 128, 256, Wp, stream);
  pool_max_kernel<<<gfor((size_t)MPTS * 256), 256, 0, stream>>>(BIG, h2, 256);
  run_bn(h2, MPTS, 256, mean, var, PAR(PP2 + 3), PAR(PP2 + 2), 2, stream);

  // ---- geoconv g1 (cin=64, hid=64, cout=128) ------------------------------
  pack_wbins_kernel<<<gfor((size_t)64 * 384), 256, 0, stream>>>(PAR(G1 + 0), Wt, 64, 64);
  run_gemm(x3, sid2, Wt, PAR(G1 + 3), nullptr, BIG, EDG, 64, 64, 384, Wp, stream);
  geo_agg_kernel<<<gfor((size_t)MPTS * 64), 256, 0, stream>>>(BIG, pceb, bidb, dnb, agg, 64);
  run_gemm(x3, nullptr, PAR(G1 + 2), PAR(G1 + 7), agg, xA, MPTS, 64, 64, 64, Wp, stream);
  run_bn(xA, MPTS, 64, mean, var, PAR(G1 + 8), PAR(G1 + 4), 1, stream);
  run_gemm(xA, nullptr, PAR(G1 + 1), PAR(G1 + 6), nullptr, l3, MPTS, 64, 64, 128, Wp, stream);
  run_bn(l3, MPTS, 128, mean, var, PAR(G1 + 9), PAR(G1 + 5), 1, stream);

  // ---- c2 (128->128) leaky BN -> l4in in xA -------------------------------
  run_gemm(l3, nullptr, PAR(C2 + 0), PAR(C2 + 1), nullptr, xA, MPTS, 128, 128, 128, Wp, stream);
  run_bn(xA, MPTS, 128, mean, var, PAR(C2 + 3), PAR(C2 + 2), 1, stream);

  // ---- geoconv g2 (cin=128, hid=64, cout=256) -----------------------------
  pack_wbins_kernel<<<gfor((size_t)128 * 384), 256, 0, stream>>>(PAR(G2 + 0), Wt, 128, 64);
  run_gemm(xA, sid2, Wt, PAR(G2 + 3), nullptr, BIG, EDG, 128, 128, 384, Wp, stream);
  geo_agg_kernel<<<gfor((size_t)MPTS * 64), 256, 0, stream>>>(BIG, pceb, bidb, dnb, agg, 64);
  run_gemm(xA, nullptr, PAR(G2 + 2), PAR(G2 + 7), agg, xB, MPTS, 128, 128, 64, Wp, stream);
  run_bn(xB, MPTS, 64, mean, var, PAR(G2 + 8), PAR(G2 + 4), 1, stream);
  run_gemm(xB, nullptr, PAR(G2 + 1), PAR(G2 + 6), nullptr, l4, MPTS, 64, 64, 256, Wp, stream);
  run_bn(l4, MPTS, 256, mean, var, PAR(G2 + 9), PAR(G2 + 5), 1, stream);

  // ---- fc = [l3 | l4 | l0_points1], f1/f2/f3 leaky BN, f4 -----------------
  concat3_kernel<<<gfor((size_t)MPTS * 640), 256, 0, stream>>>(xA, 640, l3, 128, l4, 256, h2, 256);
  run_gemm(xA, nullptr, PAR(F1 + 0), PAR(F1 + 1), nullptr, xB, MPTS, 640, 640, 512, Wp, stream);
  run_bn(xB, MPTS, 512, mean, var, PAR(F1 + 3), PAR(F1 + 2), 1, stream);
  run_gemm(xB, nullptr, PAR(F2 + 0), PAR(F2 + 1), nullptr, xA, MPTS, 512, 512, 256, Wp, stream);
  run_bn(xA, MPTS, 256, mean, var, PAR(F2 + 3), PAR(F2 + 2), 1, stream);
  run_gemm(xA, nullptr, PAR(F3 + 0), PAR(F3 + 1), nullptr, xB, MPTS, 256, 256, 128, Wp, stream);
  run_bn(xB, MPTS, 128, mean, var, PAR(F3 + 3), PAR(F3 + 2), 1, stream);
  run_gemm(xB, nullptr, PAR(F4W), PAR(F4B), nullptr, l0, MPTS, 128, 128, 64, Wp, stream);

  // ---- GRU-style update ----------------------------------------------------
  tr_bcn_kernel<<<gfor((size_t)MPTS * 64), 256, 0, stream>>>(FEA, featT, 64);
  concat3_kernel<<<gfor((size_t)MPTS * 128), 256, 0, stream>>>(xA, 128, l0, 64, featT, 64, nullptr, 0);
  run_gemm(xA, nullptr, PAR(UZ + 0), PAR(UZ + 1), nullptr, zb, MPTS, 128, 128, 64, Wp, stream);
  run_bn(zb, MPTS, 64, mean, var, PAR(UZ + 3), PAR(UZ + 2), 3, stream);   // sigmoid
  run_gemm(xA, nullptr, PAR(UR + 0), PAR(UR + 1), nullptr, rb, MPTS, 128, 128, 64, Wp, stream);
  run_bn(rb, MPTS, 64, mean, var, PAR(UR + 3), PAR(UR + 2), 3, stream);   // sigmoid
  gru_cath_kernel<<<gfor((size_t)MPTS * 128), 256, 0, stream>>>(l0, rb, featT, xB);
  run_gemm(xB, nullptr, PAR(UH + 0), PAR(UH + 1), nullptr, hhb, MPTS, 128, 128, 64, Wp, stream);
  run_bn(hhb, MPTS, 64, mean, var, PAR(UH + 3), PAR(UH + 2), 2, stream);  // relu
  gru_comb_kernel<<<gfor((size_t)MPTS * 64), 256, 0, stream>>>(zb, l0, hhb, xA); // feat2

  // ---- seg head: [feat2 | noise16] padded 80->96, s1 leaky BN, s2 ---------
  tr_bcn_kernel<<<gfor((size_t)MPTS * 16), 256, 0, stream>>>(N16, n16T, 16);
  concat3_kernel<<<gfor((size_t)MPTS * 96), 256, 0, stream>>>(xB, 96, xA, 64, n16T, 16, nullptr, 0);
  run_gemm(xB, nullptr, PAR(S1 + 0), PAR(S1 + 1), nullptr, xA, MPTS, 80, 96, 64, Wp, stream);
  run_bn(xA, MPTS, 64, mean, var, PAR(S1 + 3), PAR(S1 + 2), 1, stream);
  run_gemm(xA, nullptr, PAR(S2W), PAR(S2B), nullptr, seg, MPTS, 64, 64, 3, Wp, stream);

  // ---- outputs: (seg + pc^T, seg) -----------------------------------------
  final_out_kernel<<<gfor((size_t)BQ * 3 * NPTS), 256, 0, stream>>>(seg, PC, (float*)d_out);
}
